// QTemporal_6107443495212
// MI455X (gfx1250) — compile-verified, matrix operand order checked by reading, not run on hardware
//
#include <hip/hip_runtime.h>
#include <hip/hip_bf16.h>
#include <math.h>

// ---------------------------------------------------------------------------
// QTemporal fused pipeline for gfx1250 (MI455X)
//   proj = x @ P_w^T ; base = proj @ W_w^T + b ; LN ; scalar-gate EMA
// EMA with scalar gate g collapses to  h = sum_t g*(1-g)^(T-1-t) * LN(base_t).
// GEMMs run on v_wmma_f32_16x16x32_bf16 with 4-tile accumulation per wave
// (one shared A fragment feeds 4 independent WMMAs) and K-loop unroll-by-2
// so loads overlap matrix ops instead of draining loadcnt to 0 per WMMA.
// ---------------------------------------------------------------------------

typedef __attribute__((ext_vector_type(16))) __bf16 v16bf;
typedef __attribute__((ext_vector_type(8)))  float  v8f;

#define B_   32
#define T_   2048
#define F_   512
#define K_   1024
#define D_   512
#define TC   16               // tokens per workgroup chunk
#define CHUNKS (T_ / TC)      // 128 chunks per batch
#define XS_PITCH 528          // bf16 elems per row (1056 B, 32B-aligned, bank-skewed)
#define PS_PITCH 1056         // bf16 elems per row (2112 B)
#define BS_PITCH 528          // f32 elems per row  (2112 B) -- aliases PS region

static __device__ __forceinline__ v8f wmma_bf16(v16bf a, v16bf b, v8f c) {
    return __builtin_amdgcn_wmma_f32_16x16x32_bf16(false, a, false, b,
                                                   (short)0, c, false, false);
}

// ---------------- weight down-convert: fp32 -> bf16 -------------------------
__global__ void f32_to_bf16_kernel(const float* __restrict__ src,
                                   __bf16* __restrict__ dst, int n) {
    int i = (blockIdx.x * blockDim.x + threadIdx.x) * 4;
    if (i + 3 < n) {
        float4 v = *(const float4*)(src + i);
        dst[i + 0] = (__bf16)v.x;
        dst[i + 1] = (__bf16)v.y;
        dst[i + 2] = (__bf16)v.z;
        dst[i + 3] = (__bf16)v.w;
    }
}

// ---------------- calibration scalar: mean of x[b, :, F-8:] -----------------
__global__ void cal_kernel(const float* __restrict__ x, float* __restrict__ cal) {
    const int b = blockIdx.x, tid = threadIdx.x;
    float s = 0.f;
    const float* xb = x + (size_t)b * T_ * F_ + (F_ - 8);
    for (int t = tid; t < T_; t += 256) {
        const float* row = xb + (size_t)t * F_;
#pragma unroll
        for (int j = 0; j < 8; ++j) s += row[j];
    }
#pragma unroll
    for (int m = 16; m >= 1; m >>= 1) s += __shfl_xor(s, m, 32);
    __shared__ float wsum[8];
    const int wave = tid >> 5, lane = tid & 31;
    if (lane == 0) wsum[wave] = s;
    __syncthreads();
    if (tid == 0) {
        float tot = 0.f;
#pragma unroll
        for (int w = 0; w < 8; ++w) tot += wsum[w];
        cal[b] = tot * (1.f / (float)(T_ * 8));
    }
}

// ---------------- main fused GEMM+GEMM+LN+EMA-weight kernel -----------------
__global__ __launch_bounds__(256)
void qtemporal_main(const float* __restrict__ x,
                    const __bf16* __restrict__ Pw,     // [K,F] bf16
                    const __bf16* __restrict__ Ww,     // [D,K] bf16
                    const float* __restrict__ Wb,
                    const float* __restrict__ bias,
                    const float* __restrict__ ln_g,
                    const float* __restrict__ ln_b,
                    const float* __restrict__ alpha,
                    const float* __restrict__ th1, const float* __restrict__ ph1,
                    const float* __restrict__ th2, const float* __restrict__ ph2,
                    const float* __restrict__ w1p, const float* __restrict__ w2p,
                    const float* __restrict__ bgp,
                    const float* __restrict__ cal,
                    float* __restrict__ partials)      // [4096, D]
{
    __shared__ __align__(64) unsigned char smem[TC*XS_PITCH*2 + TC*PS_PITCH*2 + 128];
    __bf16 (*xs)[XS_PITCH] = (__bf16(*)[XS_PITCH])smem;
    __bf16 (*ps)[PS_PITCH] = (__bf16(*)[PS_PITCH])(smem + TC*XS_PITCH*2);
    float  (*bs)[BS_PITCH] = (float (*)[BS_PITCH])(smem + TC*XS_PITCH*2); // alias
    float*  stats          = (float*)(smem + TC*XS_PITCH*2 + TC*PS_PITCH*2);

    const int wg    = blockIdx.x;          // 0..4095
    const int b     = wg >> 7;
    const int t0    = (wg & 127) * TC;
    const int tid   = threadIdx.x;
    const int wave  = tid >> 5;
    const int lane  = tid & 31;
    const int lrow  = lane & 15;
    const int lhalf = (lane >> 4) << 4;    // K-half select for A/B fragments
    const int mhalf = (lane >> 4) << 3;    // M-half select for C/D tiles

    // ---- phase 0: stage x chunk into LDS as bf16 ----
    const float* xsrc = x + ((size_t)b * T_ + t0) * F_;
#pragma unroll
    for (int j = 0; j < 8; ++j) {
        int i = (tid + j * 256) * 4;                // flat f32 index < 8192
        float4 v = *(const float4*)(xsrc + i);
        int r = i >> 9, c = i & 511;
        __bf16* p = &xs[r][c];
        p[0] = (__bf16)v.x; p[1] = (__bf16)v.y;
        p[2] = (__bf16)v.z; p[3] = (__bf16)v.w;
    }
    __syncthreads();

    // ---- phase A: proj[16,1024] = xs @ Pw^T ----
    // Wave owns 8 tiles (columns wave*128 .. wave*128+127), done as 2 passes
    // of 4 tiles; one A fragment feeds 4 independent WMMAs per K-step.
#pragma unroll 1
    for (int pass = 0; pass < 2; ++pass) {
        const int nbase = wave * 128 + pass * 64;
        const __bf16* wrow = Pw + (size_t)(nbase + lrow) * F_ + lhalf;
        v8f accA[4] = {};
#pragma unroll 2
        for (int f0 = 0; f0 < F_; f0 += 32) {
            v16bf a = *(const v16bf*)&xs[lrow][f0 + lhalf];
#pragma unroll
            for (int j = 0; j < 4; ++j) {
                v16bf w = *(const v16bf*)(wrow + (size_t)j * 16 * F_ + f0);
                accA[j] = wmma_bf16(a, w, accA[j]);
            }
        }
#pragma unroll
        for (int j = 0; j < 4; ++j)
#pragma unroll
            for (int r = 0; r < 8; ++r)
                ps[r + mhalf][nbase + j * 16 + lrow] = (__bf16)accA[j][r];
    }
    __syncthreads();

    // ---- phase B: base[16,512] = ps @ Ww^T ----
    // Wave owns 4 tiles (columns wave*64 .. wave*64+63), single pass.
    const int dbase = wave * 64;
    const __bf16* wrow2 = Ww + (size_t)(dbase + lrow) * K_ + lhalf;
    v8f accB[4] = {};
#pragma unroll 2
    for (int k0 = 0; k0 < K_; k0 += 32) {
        v16bf a = *(const v16bf*)&ps[lrow][k0 + lhalf];
#pragma unroll
        for (int j = 0; j < 4; ++j) {
            v16bf w = *(const v16bf*)(wrow2 + (size_t)j * 16 * K_ + k0);
            accB[j] = wmma_bf16(a, w, accB[j]);
        }
    }
    __syncthreads();   // all ps reads done before aliased bs writes

    const float acal = alpha[0] * cal[b];
#pragma unroll
    for (int j = 0; j < 4; ++j) {
        const int col = dbase + j * 16 + lrow;
        const float add = Wb[col] + bias[col] + acal;
#pragma unroll
        for (int r = 0; r < 8; ++r)
            bs[r + mhalf][col] = accB[j][r] + add;
    }
    __syncthreads();

    // ---- LayerNorm stats: wave w owns rows 2w, 2w+1 ----
#pragma unroll
    for (int rr = 0; rr < 2; ++rr) {
        const int r = wave * 2 + rr;
        float s = 0.f, s2 = 0.f;
#pragma unroll
        for (int j = 0; j < 16; ++j) {
            float v = bs[r][lane * 16 + j];
            s += v; s2 += v * v;
        }
#pragma unroll
        for (int m = 16; m >= 1; m >>= 1) {
            s  += __shfl_xor(s,  m, 32);
            s2 += __shfl_xor(s2, m, 32);
        }
        if (lane == 0) {
            float mu  = s * (1.f / 512.f);
            float var = s2 * (1.f / 512.f) - mu * mu;
            stats[r * 2]     = mu;
            stats[r * 2 + 1] = rsqrtf(var + 1e-5f);
        }
    }
    __syncthreads();

    // ---- gate g (scalar), EMA weight w_t = g*(1-g)^(T-1-t), partial sum ----
    float g;
    {
        float z1 = cosf(th1[0]) * cosf(ph1[0]);
        float z2 = cosf(th2[0]) * cosf(ph2[0]);
        float zz = w1p[0] * z1 + w2p[0] * z2 + bgp[0];
        float sg = 1.f / (1.f + expf(-zz));
        g = fminf(fmaxf(sg, 0.05f), 0.75f);
    }
    const float lg = logf(1.f - g);

    float* out = partials + (size_t)wg * D_;
#pragma unroll
    for (int q = 0; q < 2; ++q) {
        const int d = tid + q * 256;
        const float gamma = ln_g[d], beta = ln_b[d];
        float accd = 0.f;
#pragma unroll 1
        for (int r = 0; r < TC; ++r) {
            const int t = t0 + r;
            const float w  = g * expf((float)(T_ - 1 - t) * lg);
            const float mu = stats[r * 2], rs = stats[r * 2 + 1];
            accd += w * ((bs[r][d] - mu) * rs * gamma + beta);
        }
        out[d] = accd;
    }
}

// ---------------- deterministic fixed-order chunk reduction -----------------
__global__ void finalize_kernel(const float* __restrict__ partials,
                                float* __restrict__ out) {
    const int gd = blockIdx.x * 256 + threadIdx.x;   // 0..16383
    const int b = gd >> 9, d = gd & 511;
    const float* p = partials + ((size_t)b * CHUNKS) * D_ + d;
    float s = 0.f;
#pragma unroll 1
    for (int c = 0; c < CHUNKS; ++c) s += p[(size_t)c * D_];
    out[gd] = s;
}

// ---------------------------------------------------------------------------
extern "C" void kernel_launch(void* const* d_in, const int* in_sizes, int n_in,
                              void* d_out, int out_size, void* d_ws, size_t ws_size,
                              hipStream_t stream) {
    const float* x    = (const float*)d_in[0];
    const float* P_w  = (const float*)d_in[1];
    const float* W_w  = (const float*)d_in[2];
    const float* W_b  = (const float*)d_in[3];
    const float* bias = (const float*)d_in[4];
    const float* ln_g = (const float*)d_in[5];
    const float* ln_b = (const float*)d_in[6];
    const float* alph = (const float*)d_in[7];
    const float* th1  = (const float*)d_in[8];
    const float* ph1  = (const float*)d_in[9];
    const float* th2  = (const float*)d_in[10];
    const float* ph2  = (const float*)d_in[11];
    const float* w1   = (const float*)d_in[12];
    const float* w2   = (const float*)d_in[13];
    const float* bg   = (const float*)d_in[14];

    char* ws = (char*)d_ws;
    __bf16* Pw_bf   = (__bf16*)(ws);                         // 1 MB
    __bf16* Ww_bf   = (__bf16*)(ws + (1u << 20));            // 1 MB
    float*  cal     = (float*) (ws + (2u << 20));            // 128 B
    float*  partial = (float*) (ws + (2u << 20) + 1024);     // 8 MB

    f32_to_bf16_kernel<<<512, 256, 0, stream>>>(P_w, Pw_bf, K_ * F_);
    f32_to_bf16_kernel<<<512, 256, 0, stream>>>(W_w, Ww_bf, D_ * K_);
    cal_kernel<<<B_, 256, 0, stream>>>(x, cal);
    qtemporal_main<<<B_ * CHUNKS, 256, 0, stream>>>(
        x, Pw_bf, Ww_bf, W_b, bias, ln_g, ln_b,
        alph, th1, ph1, th2, ph2, w1, w2, bg, cal, partial);
    finalize_kernel<<<(B_ * D_) / 256, 256, 0, stream>>>(partial, (float*)d_out);
}
